// Seq2Seq_76914274337316
// MI455X (gfx1250) — matrix operand order repaired and checked by reading, not compiled, not run on hardware
//
#include <hip/hip_runtime.h>
#include <hip/hip_bf16.h>

#define SEQ  4096
#define HID  512
#define G3   1536          // 3*H
#define H2   1024          // 2*H
#define NBLK 32            // decoder persistent blocks

typedef __bf16 bf16;
typedef __attribute__((ext_vector_type(16))) __bf16 v16bf;
typedef __attribute__((ext_vector_type(8)))  __bf16 v8bf;
typedef __attribute__((ext_vector_type(8)))  float  v8f;

// ---------------- math helpers (CDNA5 trans ops) ----------------
__device__ __forceinline__ float tanh_f(float x) {
    float y;
    // gfx1250 v_tanh_f32; v_nop covers the TRANS result hazard
    asm volatile("v_tanh_f32 %0, %1\n\tv_nop" : "=v"(y) : "v"(x));
    return y;
}
__device__ __forceinline__ float exp_f(float x) {
    return __builtin_amdgcn_exp2f(1.44269504f * x);
}
__device__ __forceinline__ float sigmoid_f(float x) {
    return __builtin_amdgcn_rcpf(1.0f + __builtin_amdgcn_exp2f(-1.44269504f * x));
}
__device__ __forceinline__ float wred32(float v) {
    #pragma unroll
    for (int o = 16; o > 0; o >>= 1) v += __shfl_xor(v, o, 32);
    return v;
}
__device__ __forceinline__ float dot4(float4 a, float4 b) {
    return a.x * b.x + a.y * b.y + a.z * b.z + a.w * b.w;
}

// ---------------- grid barrier (generation counter) ----------------
__device__ __forceinline__ void grid_barrier(unsigned* bar, unsigned nblk) {
    __threadfence();                 // release
    __syncthreads();
    if (threadIdx.x == 0) {
        volatile unsigned* vgen = bar + 1;
        unsigned g = *vgen;
        if (atomicAdd(bar, 1u) == nblk - 1u) {
            *(volatile unsigned*)bar = 0u;
            __threadfence();
            *vgen = g + 1u;
        } else {
            while (*vgen == g) __builtin_amdgcn_s_sleep(2);
        }
    }
    __syncthreads();
    __threadfence();                 // acquire (invalidates L0/L1)
}

// ---------------- small prep kernels ----------------
__global__ void k_init(unsigned* bar, float* ssum, float* ctx2) {
    int t = threadIdx.x;
    if (t < 2) bar[t] = 0u;
    if (t == 2) *ssum = 0.f;
    for (int i = t; i < 2 * H2; i += 256) ctx2[i] = 0.f;
}

__global__ void k_embed(const int* __restrict__ x, const float* __restrict__ emb,
                        bf16* __restrict__ xe_bf) {
    int gid = blockIdx.x * blockDim.x + threadIdx.x;   // SEQ*HID threads
    int t = gid >> 9, h = gid & (HID - 1);
    xe_bf[gid] = (bf16)emb[x[t] * HID + h];
}

__global__ void k_cast_bf16(const float* __restrict__ src, bf16* __restrict__ dst, int n) {
    int gid = blockIdx.x * blockDim.x + threadIdx.x;
    if (gid < n) dst[gid] = (bf16)src[gid];
}

// -------- WMMA GEMM: D[m,n] = sum_k A[m,k]*W[n,k] + bias[n] --------
// Each wave computes a 16x64 strip (4 N-tiles, A fragment reused 4x).
// Ping-pong software pipeline: fragment sets X (k) and Y (k+32); compute X,
// reload X at k+64, compute Y, reload Y at k+96. Register renaming does the
// rotation (no v_mov chains, no mov-induced WMMA->VALU hazard nops), and each
// load group has a full wmma group between issue and wait.
__device__ __forceinline__ v16bf load_bfrag(const bf16* p) {
    return *(const v16bf*)p;
}
__device__ __forceinline__ v16bf load_afrag(const bf16* p) {
    v8bf lo = *(const v8bf*)p;
    v8bf hi = *(const v8bf*)(p + 8);
    v16bf a;
    #pragma unroll
    for (int i = 0; i < 8; ++i) { a[i] = lo[i]; a[i + 8] = hi[i]; }
    return a;
}

__global__ void __launch_bounds__(256) k_gemm_bf16(
    const bf16* __restrict__ A, const bf16* __restrict__ W,
    const float* __restrict__ bias, float* __restrict__ D,
    int M, int N, int rev)                        // K == HID (compile-time)
{
    const int lane = threadIdx.x & 31;
    const int wave = (blockIdx.x * blockDim.x + threadIdx.x) >> 5;
    const int ngrp = N >> 6;                      // 64-wide N groups
    const int tm = wave / ngrp, tg = wave - tm * ngrp;
    if (tm * 16 >= M) return;                     // wave-uniform
    const int half = lane >> 4;                   // 0|1
    const int l15  = lane & 15;
    const int am   = tm * 16 + l15;
    // A 16x32 frag: lane<16 holds K 0..15; lane>=16 holds K 16..31
    const bf16* ap = A + (size_t)(rev ? (M - 1 - am) : am) * HID + half * 16;
    // B 32x16 frags: lane n<16 -> K 0..15 of column n, lanes 16..31 -> K 16..31
    const bf16* wp = W + (size_t)(tg * 64 + l15) * HID + half * 16;
    const size_t wstp = (size_t)16 * HID;         // next N-tile's rows

    v8f c0 = {0.f,0.f,0.f,0.f,0.f,0.f,0.f,0.f};
    v8f c1 = c0, c2 = c0, c3 = c0;

    // set X at k=0, set Y at k=32
    v16bf ax  = load_afrag(ap);
    v16bf bx0 = load_bfrag(wp);
    v16bf bx1 = load_bfrag(wp + wstp);
    v16bf bx2 = load_bfrag(wp + 2 * wstp);
    v16bf bx3 = load_bfrag(wp + 3 * wstp);
    v16bf ay  = load_afrag(ap + 32);
    v16bf by0 = load_bfrag(wp + 32);
    v16bf by1 = load_bfrag(wp + wstp + 32);
    v16bf by2 = load_bfrag(wp + 2 * wstp + 32);
    v16bf by3 = load_bfrag(wp + 3 * wstp + 32);

    #pragma unroll 1
    for (int k0 = 0; k0 < HID - 64; k0 += 64) {
        // compute X(k0), then reload X for k0+64
        c0 = __builtin_amdgcn_wmma_f32_16x16x32_bf16(false, ax, false, bx0, (short)0, c0, false, false);
        c1 = __builtin_amdgcn_wmma_f32_16x16x32_bf16(false, ax, false, bx1, (short)0, c1, false, false);
        c2 = __builtin_amdgcn_wmma_f32_16x16x32_bf16(false, ax, false, bx2, (short)0, c2, false, false);
        c3 = __builtin_amdgcn_wmma_f32_16x16x32_bf16(false, ax, false, bx3, (short)0, c3, false, false);
        ax  = load_afrag(ap + k0 + 64);
        bx0 = load_bfrag(wp + k0 + 64);
        bx1 = load_bfrag(wp + wstp + k0 + 64);
        bx2 = load_bfrag(wp + 2 * wstp + k0 + 64);
        bx3 = load_bfrag(wp + 3 * wstp + k0 + 64);
        __builtin_prefetch(ap + k0 + 384, 0, 1);   // global_prefetch_b8
        // compute Y(k0+32), then reload Y for k0+96
        c0 = __builtin_amdgcn_wmma_f32_16x16x32_bf16(false, ay, false, by0, (short)0, c0, false, false);
        c1 = __builtin_amdgcn_wmma_f32_16x16x32_bf16(false, ay, false, by1, (short)0, c1, false, false);
        c2 = __builtin_amdgcn_wmma_f32_16x16x32_bf16(false, ay, false, by2, (short)0, c2, false, false);
        c3 = __builtin_amdgcn_wmma_f32_16x16x32_bf16(false, ay, false, by3, (short)0, c3, false, false);
        ay  = load_afrag(ap + k0 + 96);
        by0 = load_bfrag(wp + k0 + 96);
        by1 = load_bfrag(wp + wstp + k0 + 96);
        by2 = load_bfrag(wp + 2 * wstp + k0 + 96);
        by3 = load_bfrag(wp + 3 * wstp + k0 + 96);
    }
    // epilogue: X(HID-64) and Y(HID-32) are already loaded
    c0 = __builtin_amdgcn_wmma_f32_16x16x32_bf16(false, ax, false, bx0, (short)0, c0, false, false);
    c1 = __builtin_amdgcn_wmma_f32_16x16x32_bf16(false, ax, false, bx1, (short)0, c1, false, false);
    c2 = __builtin_amdgcn_wmma_f32_16x16x32_bf16(false, ax, false, bx2, (short)0, c2, false, false);
    c3 = __builtin_amdgcn_wmma_f32_16x16x32_bf16(false, ax, false, bx3, (short)0, c3, false, false);
    c0 = __builtin_amdgcn_wmma_f32_16x16x32_bf16(false, ay, false, by0, (short)0, c0, false, false);
    c1 = __builtin_amdgcn_wmma_f32_16x16x32_bf16(false, ay, false, by1, (short)0, c1, false, false);
    c2 = __builtin_amdgcn_wmma_f32_16x16x32_bf16(false, ay, false, by2, (short)0, c2, false, false);
    c3 = __builtin_amdgcn_wmma_f32_16x16x32_bf16(false, ay, false, by3, (short)0, c3, false, false);

    const int mb = tm * 16 + half * 8;            // C/D: VGPR r -> M = r (+8 for hi lanes)
    #pragma unroll
    for (int q = 0; q < 4; ++q) {
        v8f c = (q == 0) ? c0 : (q == 1) ? c1 : (q == 2) ? c2 : c3;
        int n = tg * 64 + q * 16 + l15;
        float bn = bias[n];
        #pragma unroll
        for (int r = 0; r < 8; ++r)
            D[(size_t)(mb + r) * N + n] = c[r] + bn;
    }
}

// ---------------- encoder GRU scans (blockIdx.x = direction) ----------------
// Recurrent matvec uses untransposed Wh with b128 loads against b128 LDS reads.
__global__ void __launch_bounds__(1024) k_enc_scan(
    const float* __restrict__ GiAll,   // [2][SEQ][G3] precomputed Wi@x + bi
    const float* __restrict__ Wh_f,    // [G3][HID]
    const float* __restrict__ Wh_b,
    const float* __restrict__ bh_f, const float* __restrict__ bh_b,
    float* __restrict__ enc,           // [SEQ][H2]
    float* __restrict__ hcat)          // [H2]
{
    const int dir = blockIdx.x;
    const float* Gi = GiAll + (size_t)dir * SEQ * G3;
    const float* Wh = dir ? Wh_b : Wh_f;
    const float* bh = dir ? bh_b : bh_f;
    __shared__ float h[HID];
    __shared__ float gh[G3];
    const int tid = threadIdx.x;
    for (int k = tid; k < HID; k += 1024) h[k] = 0.f;
    __syncthreads();
    const float4* h4 = (const float4*)h;
    for (int t = 0; t < SEQ; ++t) {
        for (int j = tid; j < G3; j += 1024) {
            const float4* wr = (const float4*)(Wh + (size_t)j * HID);
            float s0 = 0.f, s1 = 0.f;
            #pragma unroll 4
            for (int i = 0; i < HID / 4; i += 2) {
                s0 += dot4(wr[i],     h4[i]);
                s1 += dot4(wr[i + 1], h4[i + 1]);
            }
            gh[j] = bh[j] + s0 + s1;
        }
        __syncthreads();
        if (tid < HID) {
            const float* gi = Gi + (size_t)t * G3;
            float r  = sigmoid_f(gi[tid]          + gh[tid]);
            float z  = sigmoid_f(gi[HID + tid]    + gh[HID + tid]);
            float n  = tanh_f  (gi[2 * HID + tid] + r * gh[2 * HID + tid]);
            float hn = (1.f - z) * n + z * h[tid];
            h[tid] = hn;
            size_t row = dir ? (size_t)(SEQ - 1 - t) : (size_t)t;
            enc[row * H2 + dir * HID + tid] = hn;
        }
        __syncthreads();
    }
    if (tid < HID) hcat[dir * HID + tid] = h[tid];
}

// ---------------- enc_score[t] = enc[t] . w_e ----------------
__global__ void __launch_bounds__(256) k_enc_score(
    const float* __restrict__ enc, const float* __restrict__ w_attn,
    float* __restrict__ enc_score)
{
    int w = (blockIdx.x * blockDim.x + threadIdx.x) >> 5;
    int lane = threadIdx.x & 31;
    if (w >= SEQ) return;
    const float* e  = enc + (size_t)w * H2;
    const float* we = w_attn + HID;
    float acc = 0.f;
    for (int c = lane; c < H2; c += 32) acc += e[c] * we[c];
    acc = wred32(acc);
    if (lane == 0) enc_score[w] = acc;
}

// ---------------- dh0 = W_eh @ hcat + b_eh ----------------
__global__ void __launch_bounds__(256) k_dh0(
    const float* __restrict__ W_eh, const float* __restrict__ b_eh,
    const float* __restrict__ hcat, float* __restrict__ h)
{
    int w = (blockIdx.x * blockDim.x + threadIdx.x) >> 5;
    int lane = threadIdx.x & 31;
    if (w >= HID) return;
    const float* row = W_eh + (size_t)w * H2;
    float acc = 0.f;
    for (int c = lane; c < H2; c += 32) acc += row[c] * hcat[c];
    acc = wred32(acc);
    if (lane == 0) h[w] = acc + b_eh[w];
}

// ---------------- persistent decoder: 4096 steps, 3 grid syncs/step ----------------
// h and hdot are block-local (LDS); the final gate phase is redundantly computed by
// all blocks from the shared gisum/ghsum, eliminating the 4th barrier.
__global__ void __launch_bounds__(256) k_decoder(
    const float* __restrict__ enc, const float* __restrict__ enc_score,
    const float* __restrict__ w_attn, const float* __restrict__ b_attn,
    const float* __restrict__ Wi_d,    // [G3][H2]
    const float* __restrict__ Wh_d,    // [G3][HID]
    const float* __restrict__ bi_d, const float* __restrict__ bh_d,
    const float* __restrict__ W_out, const float* __restrict__ b_out,
    const float* __restrict__ dh0,     // [HID] initial decoder state
    float* __restrict__ escore, float* __restrict__ ssum,
    float* __restrict__ ctx2,          // [2][H2], double buffered
    float* __restrict__ gisum, float* __restrict__ ghsum,
    float* __restrict__ dec_out, float* __restrict__ attns,
    unsigned* __restrict__ bar)
{
    const int tid = threadIdx.x, bid = blockIdx.x;
    const int gid = bid * 256 + tid;
    const float battn = b_attn[0];
    __shared__ float h[HID];           // block-local decoder state
    __shared__ float sctx[H2];         // block-local ctx copy
    __shared__ float smem[512];
    __shared__ float s_hdot;

    // init block-local h = dh0 and hdot = dh0 . w_d
    for (int k = tid; k < HID; k += 256) h[k] = dh0[k];
    __syncthreads();
    {
        float hd = 0.f;
        for (int k = tid; k < HID; k += 256) hd += h[k] * w_attn[k];
        smem[tid] = hd;
        __syncthreads();
        for (int st = 128; st > 0; st >>= 1) {
            if (tid < st) smem[tid] += smem[tid + st];
            __syncthreads();
        }
        if (tid == 0) s_hdot = smem[0];
        __syncthreads();
    }

    const float4* h4 = (const float4*)h;
    const float4* sc4 = (const float4*)sctx;

    for (int t = 0; t < SEQ; ++t) {
        const int p = t & 1;
        // ---- A: e_i = exp(tanh(hdot + enc_score_i + b)); sum via L2 atomics
        if (bid < 16) {                            // gid covers 0..4095
            float sc = tanh_f(s_hdot + enc_score[gid] + battn);
            float e  = exp_f(sc);
            escore[gid] = e;
            float v = wred32(e);
            if ((tid & 31) == 0) smem[tid >> 5] = v;
            __syncthreads();
            if (tid < 8) {
                float s2 = smem[tid];
                #pragma unroll
                for (int o = 4; o > 0; o >>= 1) s2 += __shfl_xor(s2, o, 8);
                if (tid == 0) atomicAdd(ssum, s2);
            }
        }
        grid_barrier(bar, NBLK);                   // B1
        // ---- B: normalize, emit attns row, accumulate ctx partials
        {
            float inv = __builtin_amdgcn_rcpf(ssum[0]);
            int i0 = bid * 128;
            if (tid < 128) {
                float a = escore[i0 + tid] * inv;
                smem[tid] = a;
                attns[(size_t)t * SEQ + i0 + tid] = a;
            }
            __syncthreads();
            float a0 = 0.f, a1 = 0.f, a2 = 0.f, a3 = 0.f;
            for (int i = 0; i < 128; ++i) {
                float a = smem[i];
                const float* e = enc + (size_t)(i0 + i) * H2;
                a0 += a * e[tid];
                a1 += a * e[tid + 256];
                a2 += a * e[tid + 512];
                a3 += a * e[tid + 768];
            }
            atomicAdd(&ctx2[p * H2 + tid      ], a0);
            atomicAdd(&ctx2[p * H2 + tid + 256], a1);
            atomicAdd(&ctx2[p * H2 + tid + 512], a2);
            atomicAdd(&ctx2[p * H2 + tid + 768], a3);
        }
        grid_barrier(bar, NBLK);                   // B2
        // ---- C: gate pre-activations (4 threads per row, blocks 0..23)
        {
            // stage ctx in LDS (b128 loads)
            ((float4*)sctx)[tid] = ((const float4*)(ctx2 + p * H2))[tid];
            __syncthreads();
            if (bid < 24) {
                int row = bid * 64 + (tid >> 2);
                int u   = tid & 3;
                const float4* wi4 = (const float4*)(Wi_d + (size_t)row * H2);
                float acc = 0.f;
                #pragma unroll 4
                for (int i = u * 64; i < u * 64 + 64; ++i) acc += dot4(wi4[i], sc4[i]);
                const float4* wh4 = (const float4*)(Wh_d + (size_t)row * HID);
                float acch = 0.f;
                #pragma unroll 4
                for (int i = u * 32; i < u * 32 + 32; ++i) acch += dot4(wh4[i], h4[i]);
                acc  += __shfl_down(acc,  2, 4);  acc  += __shfl_down(acc,  1, 4);
                acch += __shfl_down(acch, 2, 4);  acch += __shfl_down(acch, 1, 4);
                if (u == 0) { gisum[row] = acc + bi_d[row]; ghsum[row] = acch + bh_d[row]; }
            }
            if (bid == 0 && tid == 0) *ssum = 0.f; // read in B, re-written next A
        }
        grid_barrier(bar, NBLK);                   // B3
        // ---- D: gate math redundantly on ALL blocks -> block-local h, hdot
        {
            float hd = 0.f, od = 0.f;
            float hn0, hn1;
            #pragma unroll
            for (int q = 0; q < 2; ++q) {
                int k = tid + q * 256;
                float r  = sigmoid_f(gisum[k]           + ghsum[k]);
                float z  = sigmoid_f(gisum[HID + k]     + ghsum[HID + k]);
                float n  = tanh_f  (gisum[2 * HID + k]  + r * ghsum[2 * HID + k]);
                float hn = (1.f - z) * n + z * h[k];
                if (q == 0) hn0 = hn; else hn1 = hn;
                hd += hn * w_attn[k];
                od += hn * W_out[k];
            }
            __syncthreads();                       // all reads of old h done
            h[tid] = hn0; h[tid + 256] = hn1;
            smem[tid] = hd; smem[256 + tid] = od;
            if (bid == 0) {                        // zero next step's ctx buffer
                #pragma unroll
                for (int q = 0; q < 4; ++q) ctx2[(p ^ 1) * H2 + q * 256 + tid] = 0.f;
            }
            __syncthreads();
            for (int st = 128; st > 0; st >>= 1) {
                if (tid < st) { smem[tid] += smem[tid + st]; smem[256 + tid] += smem[256 + tid + st]; }
                __syncthreads();
            }
            if (tid == 0) {
                s_hdot = smem[0];
                if (bid == 0) dec_out[t] = sigmoid_f(smem[256] + b_out[0]);
            }
            __syncthreads();                       // publish s_hdot / h
        }
        // no 4th barrier: next phase A only uses block-local state + ssum (reset
        // before B3) + escore (own range, consumed before B2 of this step)
    }
}

// ---------------- host ----------------
extern "C" void kernel_launch(void* const* d_in, const int* in_sizes, int n_in,
                              void* d_out, int out_size, void* d_ws, size_t ws_size,
                              hipStream_t stream) {
    const int*   x      = (const int*)  d_in[0];
    const float* emb    = (const float*)d_in[1];
    const float* Wi_f   = (const float*)d_in[2];
    const float* Wh_f   = (const float*)d_in[3];
    const float* bi_f   = (const float*)d_in[4];
    const float* bh_f   = (const float*)d_in[5];
    const float* Wi_b   = (const float*)d_in[6];
    const float* Wh_b   = (const float*)d_in[7];
    const float* bi_b   = (const float*)d_in[8];
    const float* bh_b   = (const float*)d_in[9];
    const float* w_attn = (const float*)d_in[10];
    const float* b_attn = (const float*)d_in[11];
    const float* W_eh   = (const float*)d_in[12];
    const float* b_eh   = (const float*)d_in[13];
    const float* Wi_d   = (const float*)d_in[14];
    const float* Wh_d   = (const float*)d_in[15];
    const float* bi_d   = (const float*)d_in[16];
    const float* bh_d   = (const float*)d_in[17];
    const float* W_out  = (const float*)d_in[18];
    const float* b_out  = (const float*)d_in[19];

    char* ws = (char*)d_ws;
    size_t off = 0;
    auto alloc = [&](size_t bytes) -> void* {
        void* p = ws + off;
        off = (off + bytes + 255) & ~(size_t)255;
        return p;
    };
    bf16*  xe_bf  = (bf16*) alloc((size_t)SEQ * HID * 2);
    bf16*  wif_bf = (bf16*) alloc((size_t)G3 * HID * 2);
    bf16*  wib_bf = (bf16*) alloc((size_t)G3 * HID * 2);
    float* Gi     = (float*)alloc((size_t)2 * SEQ * G3 * 4);
    float* enc    = (float*)alloc((size_t)SEQ * H2 * 4);
    float* encsc  = (float*)alloc((size_t)SEQ * 4);
    float* hcat   = (float*)alloc((size_t)H2 * 4);
    float* hbuf   = (float*)alloc((size_t)HID * 4);
    float* escore = (float*)alloc((size_t)SEQ * 4);
    float* ssum   = (float*)alloc(4);
    float* ctx2   = (float*)alloc((size_t)2 * H2 * 4);
    float* gisum  = (float*)alloc((size_t)G3 * 4);
    float* ghsum  = (float*)alloc((size_t)G3 * 4);
    unsigned* bar = (unsigned*)alloc(8);

    float* dec_out = (float*)d_out;
    float* attns   = dec_out + SEQ;

    k_init<<<1, 256, 0, stream>>>(bar, ssum, ctx2);
    k_embed<<<(SEQ * HID) / 256, 256, 0, stream>>>(x, emb, xe_bf);
    k_cast_bf16<<<(G3 * HID) / 256, 256, 0, stream>>>(Wi_f, wif_bf, G3 * HID);
    k_cast_bf16<<<(G3 * HID) / 256, 256, 0, stream>>>(Wi_b, wib_bf, G3 * HID);

    // Input-gate GEMMs (WMMA): Gi_f = xe @ Wi_f^T + bi_f ; Gi_b = rev(xe) @ Wi_b^T + bi_b
    int gemm_blocks = (SEQ / 16) * (G3 / 64) * 32 / 256;   // 768
    k_gemm_bf16<<<gemm_blocks, 256, 0, stream>>>(xe_bf, wif_bf, bi_f, Gi,
                                                 SEQ, G3, 0);
    k_gemm_bf16<<<gemm_blocks, 256, 0, stream>>>(xe_bf, wib_bf, bi_b,
                                                 Gi + (size_t)SEQ * G3,
                                                 SEQ, G3, 1);

    k_enc_scan<<<2, 1024, 0, stream>>>(Gi, Wh_f, Wh_b, bh_f, bh_b, enc, hcat);
    k_enc_score<<<(SEQ * 32) / 256, 256, 0, stream>>>(enc, w_attn, encsc);
    k_dh0<<<(HID * 32) / 256, 256, 0, stream>>>(W_eh, b_eh, hcat, hbuf);

    k_decoder<<<NBLK, 256, 0, stream>>>(enc, encsc, w_attn, b_attn,
                                        Wi_d, Wh_d, bi_d, bh_d, W_out, b_out,
                                        hbuf, escore, ssum, ctx2,
                                        gisum, ghsum, dec_out, attns, bar);
}
